// KalmanLSTM_38233798869123
// MI455X (gfx1250) — matrix operand order repaired and checked by reading, not compiled, not run on hardware
//
#include <hip/hip_runtime.h>

typedef _Float16 v8h  __attribute__((ext_vector_type(8)));
typedef _Float16 v16h __attribute__((ext_vector_type(16)));
typedef float    v4f  __attribute__((ext_vector_type(4)));
typedef float    v8f  __attribute__((ext_vector_type(8)));

namespace {
constexpr float kDT   = 0.2f;
constexpr float kHDT2 = 0.5f * kDT * kDT;       // dt^2/2
constexpr float kGA   = kDT * kDT * kDT / 6.0f; // dt^3/6
constexpr float kGB   = 0.5f * kDT * kDT;       // dt^2/2
constexpr float kGC   = kDT;                    // dt
constexpr float kLOG2E  = 1.4426950408889634f;
constexpr float kLOG2E2 = 2.8853900817779268f;  // 2*log2(e)
}

// branch-free transcendentals: single v_exp_f32 / v_rcp_f32 TRANS ops,
// no EXEC-diverging libm paths (co-executable with WMMA on gfx1250)
__device__ __forceinline__ float fast_sigmoid(float x) {
  return __builtin_amdgcn_rcpf(1.0f + __builtin_amdgcn_exp2f(-kLOG2E * x));
}
__device__ __forceinline__ float fast_tanh(float x) {
  // tanh(x) = 1 - 2/(1 + e^(2x))
  return fmaf(-2.0f,
              __builtin_amdgcn_rcpf(1.0f + __builtin_amdgcn_exp2f(kLOG2E2 * x)),
              1.0f);
}

__device__ __forceinline__ v16h pack16(v8h lo, v8h hi) {
  return __builtin_shufflevector(lo, hi, 0,1,2,3,4,5,6,7,8,9,10,11,12,13,14,15);
}

// ---- Kalman helpers (per-lane scalar f32, exploiting F sparsity) ----
__device__ __forceinline__ void kf_fx(float X[6]) {
  X[0] = X[0] + kDT * X[1] + kHDT2 * X[2];
  X[1] = X[1] + kDT * X[2];
  X[3] = X[3] + kDT * X[4] + kHDT2 * X[5];
  X[4] = X[4] + kDT * X[5];
}

__device__ __forceinline__ void kf_fpf(float P[6][6]) {
  #pragma unroll
  for (int j = 0; j < 6; ++j) {            // P <- F P
    P[0][j] += kDT * P[1][j] + kHDT2 * P[2][j];
    P[1][j] += kDT * P[2][j];
    P[3][j] += kDT * P[4][j] + kHDT2 * P[5][j];
    P[4][j] += kDT * P[5][j];
  }
  #pragma unroll
  for (int i = 0; i < 6; ++i) {            // P <- P F^T
    P[i][0] += kDT * P[i][1] + kHDT2 * P[i][2];
    P[i][1] += kDT * P[i][2];
    P[i][3] += kDT * P[i][4] + kHDT2 * P[i][5];
    P[i][4] += kDT * P[i][5];
  }
}

// ---- one LSTM step for the 32 batch rows owned by this wave ----
// feat GEMM (zero-padded K=6) + gate GEMM, both WMMA; LSTM pointwise update
// executes entirely in C-layout registers (i/f/g/o share lane & element index
// because gate-chunk offsets are multiples of 16). Only h goes back to LDS.
__device__ __forceinline__ void lstm_step(
    int lane, int nl, int hf, int g0,
    const v16h* __restrict__ bwih,  // [8] Wih B-fragments (registers)
    const v16h* __restrict__ bwhh,  // [8] Whh B-fragments (registers)
    const float* __restrict__ bvg,  // [8] bih+bhh at this lane's column
    const v16h* __restrict__ bcf,   // [2] cfW^T B-fragments (K=6 padded)
    const float* __restrict__ bvc,  // [2] cfB at this lane's column
    _Float16* __restrict__ sA,      // [32][64] f16: cols 0-31 feat, 32-63 h
    _Float16* __restrict__ sXa,     // [32][8]  f16: X rows (K=6 padded)
    const float X[6], float* __restrict__ cc /* [2][2][8] C-layout */)
{
  // stage own X row as f16 (one b128 store)
  {
    v8h xr;
    #pragma unroll
    for (int i = 0; i < 8; ++i) xr[i] = (i < 6) ? (_Float16)X[i] : (_Float16)0.f;
    *(v8h*)&sXa[lane * 8] = xr;
  }
  __syncthreads();

  // feat = tanh(X @ cfW^T + cfB) -> f16 rows of sA[:,0:32]
  #pragma unroll
  for (int mt = 0; mt < 2; ++mt) {
    v8h xr = *(const v8h*)&sXa[(16 * mt + nl) * 8];
    v16h ax;
    #pragma unroll
    for (int i = 0; i < 16; ++i)
      ax[i] = (hf == 0 && i < 8) ? xr[i] : (_Float16)0.f;  // K6..31 = 0
    const int rb = 16 * mt + 8 * hf;
    #pragma unroll
    for (int jt = 0; jt < 2; ++jt) {
      v8f acc;
      #pragma unroll
      for (int r = 0; r < 8; ++r) acc[r] = bvc[jt];
      acc = __builtin_amdgcn_wmma_f32_16x16x32_f16(false, ax, false, bcf[jt],
                                                   (short)0, acc, false, false);
      #pragma unroll
      for (int r = 0; r < 8; ++r)
        sA[(rb + r) * 64 + jt * 16 + nl] = (_Float16)fast_tanh(acc[r]);
    }
  }
  __syncthreads();

  // gate GEMM + pointwise cell update in C layout
  #pragma unroll
  for (int mt = 0; mt < 2; ++mt) {
    const int rowa = nl + 16 * mt;      // reads rows [16mt,16mt+16)
    v16h af = pack16(*(const v8h*)&sA[rowa * 64 + g0],
                     *(const v8h*)&sA[rowa * 64 + g0 + 8]);
    v16h ah = pack16(*(const v8h*)&sA[rowa * 64 + 32 + g0],
                     *(const v8h*)&sA[rowa * 64 + 32 + g0 + 8]);
    const int rb = 16 * mt + 8 * hf;    // writes rows [16mt,16mt+16) after reads
    #pragma unroll
    for (int jt = 0; jt < 2; ++jt) {
      v8f gi, gf, gg, go;
      #pragma unroll
      for (int r = 0; r < 8; ++r) {
        gi[r] = bvg[jt];  gf[r] = bvg[jt + 2];
        gg[r] = bvg[jt + 4]; go[r] = bvg[jt + 6];
      }
      gi = __builtin_amdgcn_wmma_f32_16x16x32_f16(false, af, false, bwih[jt],     (short)0, gi, false, false);
      gi = __builtin_amdgcn_wmma_f32_16x16x32_f16(false, ah, false, bwhh[jt],     (short)0, gi, false, false);
      gf = __builtin_amdgcn_wmma_f32_16x16x32_f16(false, af, false, bwih[jt + 2], (short)0, gf, false, false);
      gf = __builtin_amdgcn_wmma_f32_16x16x32_f16(false, ah, false, bwhh[jt + 2], (short)0, gf, false, false);
      gg = __builtin_amdgcn_wmma_f32_16x16x32_f16(false, af, false, bwih[jt + 4], (short)0, gg, false, false);
      gg = __builtin_amdgcn_wmma_f32_16x16x32_f16(false, ah, false, bwhh[jt + 4], (short)0, gg, false, false);
      go = __builtin_amdgcn_wmma_f32_16x16x32_f16(false, af, false, bwih[jt + 6], (short)0, go, false, false);
      go = __builtin_amdgcn_wmma_f32_16x16x32_f16(false, ah, false, bwhh[jt + 6], (short)0, go, false, false);
      #pragma unroll
      for (int r = 0; r < 8; ++r) {
        const int ci = (mt * 2 + jt) * 8 + r;
        float c2 = fast_sigmoid(gf[r]) * cc[ci]
                 + fast_sigmoid(gi[r]) * fast_tanh(gg[r]);
        cc[ci] = c2;
        float hv = fast_sigmoid(go[r]) * fast_tanh(c2);
        sA[(rb + r) * 64 + 32 + jt * 16 + nl] = (_Float16)hv;  // h for next GEMM
      }
    }
  }
}

__global__ __launch_bounds__(32) void kalman_lstm_fused(
    const float* __restrict__ hist,
    const float* __restrict__ psx, const float* __restrict__ psy,
    const float* __restrict__ vsx, const float* __restrict__ vsy,
    const float* __restrict__ asx, const float* __restrict__ asy,
    const float* __restrict__ jerk, const float* __restrict__ coefG,
    const float* __restrict__ GR,
    const float* __restrict__ cfW, const float* __restrict__ cfB,
    const float* __restrict__ Wih, const float* __restrict__ Whh,
    const float* __restrict__ bih, const float* __restrict__ bhh,
    const float* __restrict__ coW, const float* __restrict__ coB,
    const int* __restrict__ lenp,
    float* __restrict__ out, int B)
{
  __shared__ __align__(16) _Float16 sA[32 * 64];   // [feat | h] rows, f16
  __shared__ __align__(16) _Float16 sXa[32 * 8];   // X rows, f16, K padded
  __shared__ __align__(16) float    sCmd[32 * 4];  // command shuffle buffer

  const int lane = threadIdx.x;
  const int nl = lane & 15;
  const int hf = lane >> 4;
  const int g0 = hf * 16;   // A-frag K pattern: contiguous 16 K values
  const int kb = hf * 16;   // B-frag K pattern: contiguous 16 K values

  // ---- hoist all loop-invariant B-fragments & biases into registers ----
  v16h bwih[8], bwhh[8];
  float bvg[8];
  #pragma unroll
  for (int nt = 0; nt < 8; ++nt) {
    const int n = nt * 16 + nl;           // gate column owned by this lane
    v16h w0, w1;
    #pragma unroll
    for (int i = 0; i < 16; ++i) {        // contiguous -> global b128 loads
      w0[i] = (_Float16)Wih[n * 32 + kb + i];
      w1[i] = (_Float16)Whh[n * 32 + kb + i];
    }
    bwih[nt] = w0; bwhh[nt] = w1;
    bvg[nt] = bih[n] + bhh[n];
  }
  v16h bcf[2];
  float bvc[2];
  #pragma unroll
  for (int jt = 0; jt < 2; ++jt) {
    const int n = jt * 16 + nl;           // feat column
    v16h f;
    #pragma unroll
    for (int i = 0; i < 16; ++i)
      f[i] = (hf == 0 && i < 6) ? (_Float16)cfW[n * 6 + i] : (_Float16)0.f;
    bcf[jt] = f;
    bvc[jt] = cfB[n];
  }
  v16h bco;                               // coW^T, cols 4..15 zero
  #pragma unroll
  for (int i = 0; i < 16; ++i)
    bco[i] = (nl < 4) ? (_Float16)coW[nl * 32 + kb + i] : (_Float16)0.f;
  const float bvo = (nl < 4) ? coB[nl] : 0.f;

  #pragma unroll
  for (int j = 0; j < 32; ++j) sA[lane * 64 + 32 + j] = (_Float16)0.f;  // h0 = 0
  __syncthreads();

  int b = blockIdx.x * 32 + lane;
  if (b >= B) b = B - 1;  // keep EXEC all-ones for WMMA (benign dup in tail)

  // ---- uniform model constants (accurate tanh: one-time, feeds Q model) ----
  float th[6];
  #pragma unroll
  for (int i = 0; i < 6; ++i) th[i] = tanhf(coefG[i]);
  const float gt0[6] = { kGA * th[0], kGB * th[1], kGC * th[2], 0.f, 0.f, 0.f };
  const float gt1[6] = { 0.f, 0.f, 0.f, kGA * th[3], kGB * th[4], kGC * th[5] };
  const float j0 = jerk[0], j1 = jerk[1];
  float qh0[6], qh1[6];
  #pragma unroll
  for (int i = 0; i < 6; ++i) { qh0[i] = gt0[i] * j0; qh1[i] = gt1[i] * j1; }
  const float r0 = GR[0], r1 = GR[1];
  const float R00 = r0 * r0, R01 = r0 * r1, R11 = r1 * r1;

  // ---- Kalman init (reference quirk: X[3] = v0y overwrites z0y) ----
  const float z0x = hist[(0 * B + b) * 2 + 0], z0y = hist[(0 * B + b) * 2 + 1];
  const float z1x = hist[(1 * B + b) * 2 + 0], z1y = hist[(1 * B + b) * 2 + 1];
  float X[6] = { z0x, (z1x - z0x) / kDT, 0.f, (z1y - z0y) / kDT, 0.f, 0.f };
  float P[6][6];
  #pragma unroll
  for (int i = 0; i < 6; ++i)
    #pragma unroll
    for (int jj = 0; jj < 6; ++jj) P[i][jj] = 0.f;
  P[0][0] = psx[0] * psx[0]; P[1][1] = vsx[0] * vsx[0]; P[2][2] = asx[0] * asx[0];
  P[3][3] = psy[0] * psy[0]; P[4][4] = vsy[0] * vsy[0]; P[5][5] = asy[0] * asy[0];

  float cc[32];                             // LSTM c-state in C layout
  #pragma unroll
  for (int j = 0; j < 32; ++j) cc[j] = 0.f;

  // ---------------- history filtering: 15 steps ----------------
  for (int t = 1; t < 16; ++t) {
    lstm_step(lane, nl, hf, g0, bwih, bwhh, bvg, bcf, bvc, sA, sXa, X, cc);
    kf_fx(X);
    kf_fpf(P);
    #pragma unroll
    for (int i = 0; i < 6; ++i)
      #pragma unroll
      for (int jj = 0; jj < 6; ++jj)
        P[i][jj] += qh0[i] * qh0[jj] + qh1[i] * qh1[jj];   // + Q_hist

    const float zx = hist[((size_t)t * B + b) * 2 + 0];
    const float zy = hist[((size_t)t * B + b) * 2 + 1];
    const float y0 = zx - X[0], y1 = zy - X[3];
    const float s00 = P[0][0] + R00, s01 = P[0][3] + R01;
    const float s10 = P[3][0] + R01, s11 = P[3][3] + R11;
    const float inv = 1.0f / (s00 * s11 - s01 * s10);
    const float i00 =  s11 * inv, i01 = -s01 * inv;
    const float i10 = -s10 * inv, i11 =  s00 * inv;
    float K0[6], K1[6];
    #pragma unroll
    for (int i = 0; i < 6; ++i) {
      K0[i] = P[i][0] * i00 + P[i][3] * i10;
      K1[i] = P[i][0] * i01 + P[i][3] * i11;
    }
    #pragma unroll
    for (int i = 0; i < 6; ++i) X[i] += K0[i] * y0 + K1[i] * y1;
    float M[6][6];                                   // (I-KH) P
    #pragma unroll
    for (int i = 0; i < 6; ++i)
      #pragma unroll
      for (int jj = 0; jj < 6; ++jj)
        M[i][jj] = P[i][jj] - K0[i] * P[0][jj] - K1[i] * P[3][jj];
    #pragma unroll
    for (int i = 0; i < 6; ++i) {                    // M(I-KH)^T + K R K^T
      const float u = K0[i] * R00 + K1[i] * R01;
      const float w = K0[i] * R01 + K1[i] * R11;
      #pragma unroll
      for (int jj = 0; jj < 6; ++jj)
        P[i][jj] = M[i][jj] - M[i][0] * K0[jj] - M[i][3] * K1[jj]
                 + u * K0[jj] + w * K1[jj];
    }
  }

  // ---------------- prediction rollout ----------------
  const int LP = lenp[0];
  for (int t = 0; t < LP; ++t) {
    lstm_step(lane, nl, hf, g0, bwih, bwhh, bvg, bcf, bvc, sA, sXa, X, cc);
    __syncthreads();                       // h fully written before cmd GEMM

    // command = h @ coW^T + coB via WMMA; shuffle 4 cols to owning lanes
    v8f ac[2];
    #pragma unroll
    for (int mt = 0; mt < 2; ++mt) {
      const int rowa = nl + 16 * mt;
      v16h ah = pack16(*(const v8h*)&sA[rowa * 64 + 32 + g0],
                       *(const v8h*)&sA[rowa * 64 + 32 + g0 + 8]);
      v8f a;
      #pragma unroll
      for (int r = 0; r < 8; ++r) a[r] = bvo;
      a = __builtin_amdgcn_wmma_f32_16x16x32_f16(false, ah, false, bco,
                                                 (short)0, a, false, false);
      ac[mt] = a;
    }
    if (nl < 4) {
      #pragma unroll
      for (int mt = 0; mt < 2; ++mt) {
        const int rb = 16 * mt + 8 * hf;
        #pragma unroll
        for (int r = 0; r < 8; ++r) sCmd[(rb + r) * 4 + nl] = ac[mt][r];
      }
    }
    __syncthreads();
    const v4f cv = *(const v4f*)&sCmd[lane * 4];
    const float c0 = cv[0], c1 = cv[1], c2c = cv[2], c3 = cv[3];

    kf_fx(X);
    X[0] += kGA * c0; X[1] += kGB * c0; X[2] += kGC * c0;
    X[3] += kGA * c1; X[4] += kGB * c1; X[5] += kGC * c1;
    float Gs[6];
    #pragma unroll
    for (int i = 0; i < 6; ++i) Gs[i] = gt0[i] * c2c + gt1[i] * c3;
    kf_fpf(P);
    #pragma unroll
    for (int i = 0; i < 6; ++i)
      #pragma unroll
      for (int jj = 0; jj < 6; ++jj) P[i][jj] += Gs[i] * Gs[jj];

    const float sx = sqrtf(P[0][0]);
    const float sy = sqrtf(P[3][3]);
    const float rho = (P[0][3] + P[3][0]) / (2.0f * sx * sy);
    float* o = out + ((size_t)t * B + b) * 5;
    o[0] = X[0]; o[1] = X[3]; o[2] = sx; o[3] = sy; o[4] = rho;
  }
}

extern "C" void kernel_launch(void* const* d_in, const int* in_sizes, int n_in,
                              void* d_out, int out_size, void* d_ws, size_t ws_size,
                              hipStream_t stream) {
  (void)n_in; (void)d_ws; (void)ws_size; (void)out_size;
  const float* hist  = (const float*)d_in[0];
  const float* psx   = (const float*)d_in[1];
  const float* psy   = (const float*)d_in[2];
  const float* vsx   = (const float*)d_in[3];
  const float* vsy   = (const float*)d_in[4];
  const float* asx   = (const float*)d_in[5];
  const float* asy   = (const float*)d_in[6];
  const float* jerk  = (const float*)d_in[7];
  const float* coefG = (const float*)d_in[8];
  const float* GR    = (const float*)d_in[9];
  const float* cfW   = (const float*)d_in[10];
  const float* cfB   = (const float*)d_in[11];
  const float* Wih   = (const float*)d_in[12];
  const float* Whh   = (const float*)d_in[13];
  const float* bih   = (const float*)d_in[14];
  const float* bhh   = (const float*)d_in[15];
  const float* coW   = (const float*)d_in[16];
  const float* coB   = (const float*)d_in[17];
  const int*   lenp  = (const int*)d_in[18];

  const int B = in_sizes[0] / (16 * 2);   // hist is (T=16, B, 2)
  dim3 grid((B + 31) / 32), block(32);
  kalman_lstm_fused<<<grid, block, 0, stream>>>(
      hist, psx, psy, vsx, vsy, asx, asy, jerk, coefG, GR,
      cfW, cfB, Wih, Whh, bih, bhh, coW, coB, lenp,
      (float*)d_out, B);
}
